// AttentionConvolution_1065151889993
// MI455X (gfx1250) — matrix operand order repaired and checked by reading, not compiled, not run on hardware
//
#include <hip/hip_runtime.h>

typedef __attribute__((ext_vector_type(16))) _Float16 v16h;
typedef __attribute__((ext_vector_type(8)))  _Float16 v8h;
typedef __attribute__((ext_vector_type(4)))  _Float16 v4h;
typedef __attribute__((ext_vector_type(8)))  float    v8f;

#define BATCH 16
#define NTOK  1024
#define DHID  512
#define KCAT  1024

union h16u { v16h v; v8h h2[2]; _Float16 e[16]; };

// Packed operand layout used by ALL hot-loop WMMA operands:
//   chunk[tile][lane][16 halfs], 32 B per lane, consecutive lanes contiguous
//   -> every operand load is one coalesced v16h (2x b128) per lane.
//
// A-operand lane mapping (16x32 f16 A tile): lane L: row = L&15,
//   e[0..7]  = K 8*(L>>4)+{0..7},  e[8..15] = K 8*(L>>4)+16+{0..7}
// B-operand lane mapping (32x16 f16 B tile): lane L: col = L&15,
//   e[0..7]  = K 8*(L>>4)+{0..7},  e[8..15] = K 8*(L>>4)+16+{0..7}

// ---------------------------------------------------------------------------
// Pack hid [B,N,D] f32 -> hidB chunks for the aggregation GEMM B operand.
// chunk id c = ((b*32 + s)*32 + dt)*32 + L ; value = hid[b][s*32+8*(L>>4)+(i&7)+16*(i>>3)][dt*16+(L&15)]
// ---------------------------------------------------------------------------
__global__ __launch_bounds__(256)
void k_pack_hid(const float* __restrict__ hid, _Float16* __restrict__ hidB) {
    int c  = blockIdx.x * 256 + threadIdx.x;
    int L  = c & 31;
    int dt = (c >> 5) & 31;
    int s  = (c >> 10) & 31;
    int b  = c >> 15;
    int d  = dt * 16 + (L & 15);
    int mb = s * 32 + ((L >> 4) << 3);
    const float* src = hid + ((size_t)b * NTOK + mb) * DHID + d;
    h16u o;
    #pragma unroll
    for (int i = 0; i < 16; ++i) o.e[i] = (_Float16)src[(size_t)((i & 7) + ((i & 8) << 1)) * DHID];
    *(v16h*)(hidB + (size_t)c * 16) = o.v;
}

// ---------------------------------------------------------------------------
// Pack W [KCAT, D] f32 -> WB chunks for the projection GEMM B operand.
// chunk id c = (s*32 + dt)*32 + L ; value = W[s*32+8*(L>>4)+(i&7)+16*(i>>3)][dt*16+(L&15)]
// ---------------------------------------------------------------------------
__global__ __launch_bounds__(256)
void k_pack_W(const float* __restrict__ W, _Float16* __restrict__ WB) {
    int c  = blockIdx.x * 256 + threadIdx.x;
    int L  = c & 31;
    int dt = (c >> 5) & 31;
    int s  = c >> 10;
    int d  = dt * 16 + (L & 15);
    int kb = s * 32 + ((L >> 4) << 3);
    const float* src = W + (size_t)kb * DHID + d;
    h16u o;
    #pragma unroll
    for (int i = 0; i < 16; ++i) o.e[i] = (_Float16)src[(size_t)((i & 7) + ((i & 8) << 1)) * DHID];
    *(v16h*)(WB + (size_t)c * 16) = o.v;
}

// Scatter one f32 into the packed A-operand buffer for the projection GEMM.
__device__ __forceinline__
void store_aggA(_Float16* aggA, int blk, int row, int k, float v) {
    int s  = k >> 5, kl = k & 31;
    int L  = row + (((kl >> 3) & 1) << 4);
    int j  = (kl & 7) | ((kl & 16) >> 1);
    aggA[((((size_t)blk * 32 + s) * 32 + L) << 4) + j] = (_Float16)v;
}

// ---------------------------------------------------------------------------
// Aggregation: agg[b,n, j*512+d] = (1/cnt_j) * sum_{m: adj==j+1} hid[b,m,d]
// One block per (b, 16-row n-tile); 8 waves x (2 dirs x 4 d-tiles).
// Dynamic LDS: [0,64K)   adj row-tile (async-staged)
//              [64K,128K) packed 0/1 f16 mask A-tiles msk[dir][s][lane] (32B chunks)
//              [128K,..)  float cnt[2][16]
// ---------------------------------------------------------------------------
__global__ __launch_bounds__(256)
void k_aggregate(const int* __restrict__ adj, const _Float16* __restrict__ hidB,
                 _Float16* __restrict__ aggA) {
    extern __shared__ char smem[];
    int*   adjT = (int*)smem;                       // 64 KB
    v16h*  msk  = (v16h*)(smem + 65536);            // [2][32][32] 32B chunks = 64 KB
    float* cntF = (float*)(smem + 131072);          // [2][16]

    int blk = blockIdx.x;
    int b   = blk >> 6;
    int n0  = (blk & 63) << 4;
    int tid = threadIdx.x, wave = tid >> 5, lane = tid & 31;

    if (tid < 32) cntF[tid] = 0.0f;

    // ---- Phase 1: async-copy the contiguous 64 KB adj row-tile into LDS ----
    {
        const char* g    = (const char*)(adj + ((size_t)b * NTOK + n0) * NTOK);
        unsigned    lds0 = (unsigned)(uintptr_t)adjT;   // low 32 bits = LDS offset
        #pragma unroll
        for (int i = 0; i < 16; ++i) {
            unsigned    off  = (unsigned)(tid * 16 + i * 4096);
            unsigned    ldsa = lds0 + off;
            const char* ga   = g + off;
            asm volatile("global_load_async_to_lds_b128 %0, %1, off"
                         :: "v"(ldsa), "v"(ga) : "memory");
        }
        asm volatile("s_wait_asynccnt 0" ::: "memory");
    }
    __syncthreads();

    // ---- Phase 2: build packed 0/1 mask tiles ONCE per block + exact counts ----
    const _Float16 H1 = (_Float16)1.0f, H0 = (_Float16)0.0f;
    #pragma unroll 1
    for (int row = 0; row < 16; ++row) {
        int4 q = ((const int4*)adjT)[row * 256 + tid];
        int k0 = tid * 4;                  // 4 consecutive K inside one 8-run
        int kl = k0 & 31;
        int s  = k0 >> 5;
        int Lf = row + (((kl >> 3) & 1) << 4);
        int j0 = (kl & 7) | ((kl & 16) >> 1);       // 0,4,8,12
        int vv[4] = {q.x, q.y, q.z, q.w};
        v4h p0, p1;
        int c0 = 0, c1 = 0;
        #pragma unroll
        for (int i = 0; i < 4; ++i) {
            p0[i] = (vv[i] == 1) ? H1 : H0;
            p1[i] = (vv[i] == 2) ? H1 : H0;
            c0 += (vv[i] == 1);
            c1 += (vv[i] == 2);
        }
        *(v4h*)((_Float16*)&msk[(0 * 32 + s) * 32 + Lf] + j0) = p0;
        *(v4h*)((_Float16*)&msk[(1 * 32 + s) * 32 + Lf] + j0) = p1;
        #pragma unroll
        for (int off = 16; off >= 1; off >>= 1) {
            c0 += __shfl_xor(c0, off, 32);
            c1 += __shfl_xor(c1, off, 32);
        }
        if (lane == 0) {
            atomicAdd(&cntF[row],      (float)c0);
            atomicAdd(&cntF[16 + row], (float)c1);
        }
    }
    __syncthreads();

    // ---- Phase 3: hot loop: 2 LDS mask loads + 4 coalesced B loads + 8 WMMA ----
    const int d0 = wave * 64;
    v8f acc[2][4];
    #pragma unroll
    for (int j = 0; j < 2; ++j)
        #pragma unroll
        for (int t = 0; t < 4; ++t) acc[j][t] = (v8f)(0.0f);

    for (int s = 0; s < 32; ++s) {
        v16h a0 = msk[(0 * 32 + s) * 32 + lane];
        v16h a1 = msk[(1 * 32 + s) * 32 + lane];
        #pragma unroll
        for (int t = 0; t < 4; ++t) {
            size_t ci = (((size_t)b * 32 + s) * 32 + (wave * 4 + t)) * 32 + lane;
            const _Float16* bp = hidB + ci * 16;
            __builtin_prefetch(bp + 32 * 32 * 16, 0, 1);   // next K-step
            v16h bt = *(const v16h*)bp;
            acc[0][t] = __builtin_amdgcn_wmma_f32_16x16x32_f16(
                false, a0, false, bt, (short)0, acc[0][t], false, false);
            acc[1][t] = __builtin_amdgcn_wmma_f32_16x16x32_f16(
                false, a1, false, bt, (short)0, acc[1][t], false, false);
        }
    }

    // ---- Epilogue: scale by 1/(cnt+eps), scatter into packed A buffer ----
    #pragma unroll
    for (int t = 0; t < 4; ++t) {
        int dcol = d0 + t * 16 + (lane & 15);
        #pragma unroll
        for (int r = 0; r < 8; ++r) {
            int orow = r + ((lane >> 4) << 3);
            float sc0 = 1.0f / (cntF[orow] + 1e-13f);
            float sc1 = 1.0f / (cntF[16 + orow] + 1e-13f);
            store_aggA(aggA, blk, orow, dcol,        acc[0][t][r] * sc0);
            store_aggA(aggA, blk, orow, DHID + dcol, acc[1][t][r] * sc1);
        }
    }
}

// ---------------------------------------------------------------------------
// Projection + bias + ReLU + residual + LayerNorm (fused).
// One block per (b, 16-row n-tile); 8 waves x 4 d-tiles cover all 512 cols.
// ---------------------------------------------------------------------------
__global__ __launch_bounds__(256)
void k_project_ln(const _Float16* __restrict__ aggA, const _Float16* __restrict__ WB,
                  const float* __restrict__ hid, const float* __restrict__ bias,
                  const float* __restrict__ gamma, const float* __restrict__ beta,
                  float* __restrict__ out) {
    __shared__ float tile[16 * DHID];                // 32 KB

    int blk = blockIdx.x;
    int b   = blk >> 6;
    int n0  = (blk & 63) << 4;
    int tid = threadIdx.x, wave = tid >> 5, lane = tid & 31;
    const int d0 = wave * 64;

    v8f acc[4];
    #pragma unroll
    for (int t = 0; t < 4; ++t) acc[t] = (v8f)(0.0f);

    const v16h* Ach = (const v16h*)aggA + (size_t)blk * 32 * 32;
    const v16h* Bch = (const v16h*)WB;

    for (int s = 0; s < 32; ++s) {
        const v16h* ap = Ach + s * 32 + lane;
        __builtin_prefetch(ap + 32, 0, 1);
        v16h a = *ap;
        #pragma unroll
        for (int t = 0; t < 4; ++t) {
            v16h bt = Bch[((size_t)s * 32 + (wave * 4 + t)) * 32 + lane];
            acc[t] = __builtin_amdgcn_wmma_f32_16x16x32_f16(
                false, a, false, bt, (short)0, acc[t], false, false);
        }
    }

    // bias + ReLU + residual -> LDS tile
    #pragma unroll
    for (int t = 0; t < 4; ++t) {
        int ocol = d0 + t * 16 + (lane & 15);
        float bb = bias[ocol];
        #pragma unroll
        for (int r = 0; r < 8; ++r) {
            int orow = r + ((lane >> 4) << 3);
            float x = acc[t][r] + bb;
            x = x > 0.0f ? x : 0.0f;
            x += hid[((size_t)b * NTOK + n0 + orow) * DHID + ocol];
            tile[orow * DHID + ocol] = x;
        }
    }
    __syncthreads();

    // LayerNorm: wave w normalizes rows 2w, 2w+1 (wave32 shuffle reduction)
    #pragma unroll
    for (int rr = 0; rr < 2; ++rr) {
        int row = wave * 2 + rr;
        float s = 0.0f;
        #pragma unroll
        for (int i = 0; i < 16; ++i) s += tile[row * DHID + i * 32 + lane];
        #pragma unroll
        for (int off = 16; off >= 1; off >>= 1) s += __shfl_xor(s, off, 32);
        float mu = s * (1.0f / (float)DHID);

        float vs = 0.0f;
        #pragma unroll
        for (int i = 0; i < 16; ++i) {
            float d = tile[row * DHID + i * 32 + lane] - mu;
            vs += d * d;
        }
        #pragma unroll
        for (int off = 16; off >= 1; off >>= 1) vs += __shfl_xor(vs, off, 32);
        float inv = rsqrtf(vs * (1.0f / (float)DHID) + 1e-5f);

        size_t obase = ((size_t)b * NTOK + n0 + row) * DHID;
        #pragma unroll
        for (int i = 0; i < 16; ++i) {
            int col = i * 32 + lane;
            float x = tile[row * DHID + col];
            out[obase + col] = (x - mu) * inv * gamma[col] + beta[col];
        }
    }
}

// ---------------------------------------------------------------------------
extern "C" void kernel_launch(void* const* d_in, const int* in_sizes, int n_in,
                              void* d_out, int out_size, void* d_ws, size_t ws_size,
                              hipStream_t stream) {
    const int*   adj   = (const int*)d_in[0];
    const float* hid   = (const float*)d_in[1];
    const float* W     = (const float*)d_in[2];
    const float* bias  = (const float*)d_in[3];
    const float* gamma = (const float*)d_in[4];
    const float* beta  = (const float*)d_in[5];
    float* out = (float*)d_out;

    char* ws = (char*)d_ws;
    _Float16* hidB = (_Float16*)ws;                                  // 16 MB
    _Float16* WB   = (_Float16*)(ws + (size_t)16 * 1024 * 1024);     //  1 MB
    _Float16* aggA = (_Float16*)(ws + (size_t)17 * 1024 * 1024);     // 32 MB

    k_pack_hid<<<(BATCH * 32 * 32 * 32) / 256, 256, 0, stream>>>(hid, hidB);
    k_pack_W<<<(32 * 32 * 32) / 256, 256, 0, stream>>>(W, WB);

    size_t ldsBytes = 65536 + 65536 + 2 * 16 * sizeof(float);
    k_aggregate<<<BATCH * (NTOK / 16), 256, ldsBytes, stream>>>(adj, hidB, aggA);
    k_project_ln<<<BATCH * (NTOK / 16), 256, 0, stream>>>(aggA, WB, hid, bias,
                                                          gamma, beta, out);
}
